// LSTM_84593675862765
// MI455X (gfx1250) — matrix operand order repaired and checked by reading, not compiled, not run on hardware
//
#include <hip/hip_runtime.h>
#include <hip/hip_bf16.h>
#include <stdint.h>

// ---------------------------------------------------------------------------
// Haste LSTM forward, MI455X (gfx1250, wave32, WMMA + TDM).
//   T=512, B=64, I=H=1024, gates (i,g,f,o).
//
//  * prep: W,R -> bf16 pre-swizzled WMMA B-fragments; x -> bf16 row-major.
//  * persistent kernel: 64 blocks x 512 threads (16 waves). Block cb owns
//    hidden cols [16cb,16cb+16) for all 4 gates. R fragments (128 KB) pinned
//    in LDS all 512 steps (TDM-loaded once). Each step, wave 0 TDM-stages
//    h_prev (128 KB) into LDS overlapped with the x@W half of the K loop
//    (TENSORcnt), then h@R reads A-frags via ds_load_b128.
//  * c state lives in LDS for the whole run; h double-buffered bf16 in ws.
//  * device-wide step barrier: monotonic atomic counter + s_sleep spin.
// ---------------------------------------------------------------------------

#define T_STEPS 512
#define BATCH   64
#define IDIM    1024
#define HDIM    1024
#define NCB     64
#define KC      32
#define THREADS 512

typedef __attribute__((ext_vector_type(16))) __bf16 v16bf;
typedef __attribute__((ext_vector_type(8)))  float  v8f;
typedef __attribute__((ext_vector_type(4)))  unsigned v4u;
typedef __attribute__((ext_vector_type(8)))  int      v8i_;
typedef __attribute__((ext_vector_type(4)))  int      v4i_;

union BFrag { v16bf v; uint4 q[2]; };
union AFrag { v16bf v; uint4 q[2]; };

__device__ __forceinline__ unsigned short f2bf(float f) {
  union { float f; unsigned u; } a; a.f = f;
  return (unsigned short)((a.u + 0x7fffu + ((a.u >> 16) & 1u)) >> 16);
}
__device__ __forceinline__ unsigned pk_bf16(float lo, float hi) {
  union { float f; unsigned u; } a, b; a.f = lo; b.f = hi;
  unsigned ul = (a.u + 0x7fffu + ((a.u >> 16) & 1u)) >> 16;
  unsigned uh = (b.u + 0x7fffu + ((b.u >> 16) & 1u)) >> 16;
  return ul | (uh << 16);
}

// 1-D TDM copy: 16384 x 8B = 128 KB, global -> LDS. (6-arg clang-23 builtin.)
__device__ __forceinline__ void tdm_copy_128k(const void* src, unsigned lds_off) {
  unsigned long long ga = (unsigned long long)(uintptr_t)src;
  v4u g0 = {};
  g0[0] = 1u;                                   // count = 1 (valid descriptor)
  g0[1] = lds_off;                              // lds_addr (bytes)
  g0[2] = (unsigned)(ga & 0xffffffffu);         // global_addr[31:0]
  g0[3] = (unsigned)((ga >> 32) & 0x1ffffffu)   // global_addr[56:32]
        | (2u << 30);                           // type = 2 ("image")
  v8i_ g1 = {};
  g1[0] = (int)(3u << 16);                      // data_size = 8 bytes
  g1[1] = (int)(16384u << 16);                  // tensor_dim0[15:0] (bits 63:48)
  g1[2] = 0;                                    // tensor_dim0[31:16] = 0
  g1[3] = (int)(16384u << 16);                  // tile_dim0 (bits 127:112)
  g1[4] = 0;                                    // tile_dim1/2 = 0 (1-D tile)
  g1[5] = 16384;                                // tensor_dim0_stride[31:0]
  g1[6] = 0;
  g1[7] = 0;
  v4i_ z4 = {};
  v8i_ z8 = {};
  __builtin_amdgcn_tensor_load_to_lds(g0, g1, z4, z4, z8, 0);
}

// ---------------------------------------------------------------------------
// Pre-swizzle W,R into bf16 WMMA B-fragments:
//   dst[cb][g][kc][lane][jj] = M[kc*32 + (lane>>4)*16 + jj][g*1024+cb*16+(lane&15)]
// ---------------------------------------------------------------------------
__global__ void lstm_prep_weights(const float* __restrict__ Wk,
                                  const float* __restrict__ Rk,
                                  unsigned short* __restrict__ wsW,
                                  unsigned short* __restrict__ wsR) {
  unsigned idx = blockIdx.x * blockDim.x + threadIdx.x;   // 0 .. 2*262144-1
  unsigned m   = idx >> 18;
  unsigned r   = idx & 0x3ffffu;
  unsigned g   = (r >> 10) & 3u;
  unsigned cb  = r >> 12;
  unsigned kc  = (r >> 5) & 31u;
  unsigned ln  = r & 31u;
  unsigned hi  = (ln >> 4) << 4;
  unsigned col = g * 1024u + cb * 16u + (ln & 15u);
  unsigned k0  = kc * 32u;
  const float* src = m ? Rk : Wk;
  unsigned short* dst = (m ? wsR : wsW) + (size_t)r * 16u;
#pragma unroll
  for (unsigned jj = 0; jj < 16; ++jj)
    dst[jj] = f2bf(src[(size_t)(k0 + hi + jj) * 4096u + col]);
}

// x (fp32, [T][B][I]) -> bf16 row-major, 8 elements/thread.
__global__ void lstm_prep_x(const float* __restrict__ x,
                            unsigned short* __restrict__ xbf) {
  size_t i = ((size_t)blockIdx.x * blockDim.x + threadIdx.x) * 8u;
  float4 f0 = *(const float4*)(x + i);
  float4 f1 = *(const float4*)(x + i + 4u);
  uint4 q;
  q.x = pk_bf16(f0.x, f0.y); q.y = pk_bf16(f0.z, f0.w);
  q.z = pk_bf16(f1.x, f1.y); q.w = pk_bf16(f1.z, f1.w);
  *(uint4*)(xbf + i) = q;
}

__global__ void lstm_prep_zero(unsigned* __restrict__ h0u,
                               unsigned* __restrict__ cnt) {
  unsigned idx = blockIdx.x * blockDim.x + threadIdx.x;
  if (idx < 32768u) h0u[idx] = 0u;
  if (idx == 0u)    *cnt = 0u;
}

// ---------------------------------------------------------------------------
// Persistent LSTM kernel. grid = 64 blocks, block = 512 threads (16 waves).
// ---------------------------------------------------------------------------
__launch_bounds__(THREADS, 1)
__global__ void lstm_persistent(const unsigned short* __restrict__ xbf,
                                const float* __restrict__ bias,
                                const unsigned short* __restrict__ wsW,
                                const unsigned short* __restrict__ wsR,
                                unsigned short* __restrict__ h0,
                                unsigned short* __restrict__ h1,
                                unsigned* __restrict__ barrier_cnt,
                                float* __restrict__ out) {
  __shared__ uint4 Rlds[4][KC][32][2];       // 128 KB: resident R fragments
  __shared__ uint4 Hlds[BATCH][HDIM / 8];    // 128 KB: TDM-staged h_prev (bf16)
  __shared__ float stage[4][4][16][16];      //  16 KB: gate pre-activations
  __shared__ float cst[BATCH][16];           //   4 KB: persistent cell state

  const unsigned tid  = threadIdx.x;
  const unsigned lane = tid & 31u;
  const unsigned wv   = tid >> 5;
  const unsigned mt   = wv & 3u;
  const unsigned g    = wv >> 2;
  const unsigned cb   = blockIdx.x;
  const unsigned n0   = cb * 16u;

  const unsigned rlds_off = (unsigned)(uintptr_t)(void*)&Rlds[0][0][0][0];
  const unsigned hlds_off = (unsigned)(uintptr_t)(void*)&Hlds[0][0];

  // One-time: TDM this block's R fragment slice (128 KB) into LDS; zero c.
  if (wv == 0)
    tdm_copy_128k(wsR + (size_t)cb * (4u * KC * 32u * 16u), rlds_off);
  for (unsigned i = tid; i < BATCH * 16u; i += THREADS) (&cst[0][0])[i] = 0.f;
  if (wv == 0) __builtin_amdgcn_s_wait_tensorcnt(0);
  __syncthreads();

  const uint4* wsl = (const uint4*)wsW +
                     ((size_t)cb * 4u + g) * (KC * 32u * 2u) + (size_t)lane * 2u;

  const unsigned m0   = mt * 16u;
  const unsigned row  = m0 + (lane & 15u);        // batch row this lane feeds A
  const unsigned hi8  = (lane >> 4) << 4;         // 0 or 16 (A-frag K split)
  const unsigned mrow = (lane >> 4) ? 8u : 0u;    // C-frag row base

  for (int t = 0; t < T_STEPS; ++t) {
    const unsigned short* hrd = (t & 1) ? h1 : h0;
    unsigned short*       hwr = (t & 1) ? h0 : h1;
    const unsigned short* xrow = xbf + (size_t)t * (BATCH * IDIM)
                                     + (size_t)row * IDIM;

    // Kick TDM stage of h_prev into LDS; overlaps with the x@W half below.
    if (wv == 0) tdm_copy_128k(hrd, hlds_off);

    // Prefetch next timestep's x rows into GL2 while we compute.
    if (t + 1 < T_STEPS) {
      const unsigned short* nx = xrow + (size_t)(BATCH * IDIM);
      __builtin_prefetch(nx, 0, 1);
      __builtin_prefetch(nx + 512u, 0, 1);
    }

    v8f acc = {};

    // ---- x_t @ W  (K = 1024; pure b128 loads, no packing in serial loop) --
#pragma unroll 4
    for (unsigned kc = 0; kc < KC; ++kc) {
      const unsigned k0 = kc * 32u;
      AFrag a;
      const uint4* xp = (const uint4*)(xrow + k0 + hi8);
      a.q[0] = xp[0];                 // K k0+hi8 .. +7
      a.q[1] = xp[1];                 // K k0+hi8+8 .. +15
      BFrag b;
      b.q[0] = wsl[(size_t)kc * 64u];
      b.q[1] = wsl[(size_t)kc * 64u + 1u];
      acc = __builtin_amdgcn_wmma_f32_16x16x32_bf16(
          false, a.v, false, b.v, (short)0, acc, false, false);
    }

    // Wait for the TDM h stage, then make it visible block-wide.
    if (wv == 0) __builtin_amdgcn_s_wait_tensorcnt(0);
    __syncthreads();

    // ---- h_{t-1} @ R  (K = 1024; A from LDS h tile, B from LDS R frags) ---
#pragma unroll 4
    for (unsigned kc = 0; kc < KC; ++kc) {
      const unsigned k0 = kc * 32u;
      AFrag a;
      const uint4* hp = &Hlds[row][(k0 + hi8) >> 3];
      a.q[0] = hp[0];
      a.q[1] = hp[1];
      BFrag b;
      b.q[0] = Rlds[g][kc][lane][0];
      b.q[1] = Rlds[g][kc][lane][1];
      acc = __builtin_amdgcn_wmma_f32_16x16x32_bf16(
          false, a.v, false, b.v, (short)0, acc, false, false);
    }

    // ---- spill gate tile to LDS stage (C-frag: VGPR r -> M = mrow+r) ------
#pragma unroll
    for (int r2 = 0; r2 < 8; ++r2)
      stage[mt][g][mrow + (unsigned)r2][lane & 15u] = acc[r2];
    __syncthreads();

    // ---- elementwise LSTM update (block owns 16 hidden cols, 64 rows) ----
    for (unsigned e = tid; e < BATCH * 16u; e += THREADS) {
      const unsigned b_  = e >> 4, j = e & 15u;
      const unsigned mt2 = b_ >> 4, ml = b_ & 15u;
      const float vi = stage[mt2][0][ml][j] + bias[0u * 1024u + n0 + j];
      const float vg = stage[mt2][1][ml][j] + bias[1u * 1024u + n0 + j];
      const float vf = stage[mt2][2][ml][j] + bias[2u * 1024u + n0 + j];
      const float vo = stage[mt2][3][ml][j] + bias[3u * 1024u + n0 + j];
      const float ig = 1.f / (1.f + __expf(-vi));
      const float gg = tanhf(vg);
      const float fg = 1.f / (1.f + __expf(-vf));
      const float og = 1.f / (1.f + __expf(-vo));
      const float cn = fg * cst[b_][j] + ig * gg;
      cst[b_][j] = cn;
      const float hn = og * tanhf(cn);
      out[((size_t)t * BATCH + b_) * HDIM + n0 + j] = hn;
      hwr[(size_t)b_ * HDIM + n0 + j] = f2bf(hn);
      if (t == T_STEPS - 1) {
        const size_t base = (size_t)T_STEPS * BATCH * HDIM;
        out[base + (size_t)b_ * HDIM + n0 + j] = hn;                        // h_n
        out[base + (size_t)BATCH * HDIM + (size_t)b_ * HDIM + n0 + j] = cn; // c_n
      }
    }

    // ---- device-wide step barrier (monotonic counter) ----
    __syncthreads();
    if (tid == 0) {
      __threadfence();
      __hip_atomic_fetch_add(barrier_cnt, 1u, __ATOMIC_ACQ_REL,
                             __HIP_MEMORY_SCOPE_AGENT);
      const unsigned target = (unsigned)(NCB * (t + 1));
      while (__hip_atomic_load(barrier_cnt, __ATOMIC_ACQUIRE,
                               __HIP_MEMORY_SCOPE_AGENT) < target)
        __builtin_amdgcn_s_sleep(2);
    }
    __syncthreads();
  }
}

// ---------------------------------------------------------------------------
// Host entry. ws layout:
//   [0, 8MB)            wsW bf16 fragments
//   [8MB, 16MB)         wsR bf16 fragments
//   [16MB, 80MB)        x as bf16, row-major [T][B][I]
//   [80MB, +128KB)      h buffer 0 (bf16)
//   [+128KB, +256KB)    h buffer 1 (bf16)
//   [+256KB]            barrier counter
// ---------------------------------------------------------------------------
extern "C" void kernel_launch(void* const* d_in, const int* in_sizes, int n_in,
                              void* d_out, int out_size, void* d_ws, size_t ws_size,
                              hipStream_t stream) {
  (void)in_sizes; (void)n_in; (void)out_size; (void)ws_size;
  const float* x  = (const float*)d_in[0];
  const float* Wk = (const float*)d_in[1];
  const float* Rk = (const float*)d_in[2];
  const float* bs = (const float*)d_in[3];
  float* out = (float*)d_out;

  char* ws = (char*)d_ws;
  unsigned short* wsW = (unsigned short*)(ws);
  unsigned short* wsR = (unsigned short*)(ws + (size_t)(8u << 20));
  unsigned short* xbf = (unsigned short*)(ws + (size_t)(16u << 20));
  const size_t hoff   = (size_t)(16u << 20) + (size_t)T_STEPS * BATCH * IDIM * 2u;
  unsigned short* h0  = (unsigned short*)(ws + hoff);
  unsigned short* h1  = (unsigned short*)(ws + hoff + 131072u);
  unsigned* cnt       = (unsigned*)(ws + hoff + 262144u);

  lstm_prep_weights<<<2048, 256, 0, stream>>>(Wk, Rk, wsW, wsR);
  lstm_prep_x<<<16384, 256, 0, stream>>>(x, xbf);
  lstm_prep_zero<<<128, 256, 0, stream>>>((unsigned*)h0, cnt);
  lstm_persistent<<<NCB, THREADS, 0, stream>>>(xbf, bs, wsW, wsR, h0, h1, cnt, out);
}
